// RNNSubnet_14181982012197
// MI455X (gfx1250) — matrix operand order repaired and checked by reading, not compiled, not run on hardware
//
#include <hip/hip_runtime.h>
#include <hip/hip_bf16.h>
#include <math.h>

// ---------------------------------------------------------------------------
// LayerNorm-LSTM (2 layers, T=8192, B=1, I=H=1024) + FC/ReLU for MI455X.
//  * Time-parallel GEMMs (igates, FC): bf16 WMMA, f32 accumulate,
//    b128 LDS fragment loads (A row-major, B staged transposed).
//  * Recurrence: persistent kernel, w_hh slice resident in LDS (256KB/WGP,
//    CDNA5-only), 1024 thr/WG, 2 grid barriers per timestep.
// ---------------------------------------------------------------------------

typedef __bf16 bf16_t;
typedef __attribute__((ext_vector_type(4)))  __bf16 v4bf;
typedef __attribute__((ext_vector_type(8)))  __bf16 v8bf;
typedef __attribute__((ext_vector_type(16))) __bf16 v16bf;
typedef __attribute__((ext_vector_type(8)))  float  v8f;

#define EPSLN 1e-5f

// ---------------------------------------------------------------------------
// Block-wide sum; works for 8..32 waves (wave32)
// ---------------------------------------------------------------------------
__device__ __forceinline__ float block_sum(float v, float* red) {
#pragma unroll
  for (int o = 16; o > 0; o >>= 1) v += __shfl_xor(v, o, 32);
  __syncthreads();  // protect red against back-to-back calls
  const int wv = threadIdx.x >> 5, ln = threadIdx.x & 31;
  if (ln == 0) red[wv] = v;
  __syncthreads();
  if (wv == 0) {
    v = (ln < (int)(blockDim.x >> 5)) ? red[ln] : 0.0f;
#pragma unroll
    for (int o = 16; o > 0; o >>= 1) v += __shfl_xor(v, o, 32);
    if (ln == 0) red[0] = v;
  }
  __syncthreads();
  v = red[0];
  return v;
}

__device__ __forceinline__ float sigm(float x) {
  return 1.0f / (1.0f + __expf(-x));
}

__device__ __forceinline__ v16bf cat8(v8bf lo, v8bf hi) {
  return __builtin_shufflevector(lo, hi, 0, 1, 2, 3, 4, 5, 6, 7,
                                 8, 9, 10, 11, 12, 13, 14, 15);
}

// ---------------------------------------------------------------------------
// WMMA GEMM:  C[M,N] = A[M,K] * W[N,K]^T  (+bias, +relu)
// Block = 256 thr = 8 waves (4M x 2N); block tile 64x64; each wave computes
// one 16x16 A-frag against two B-frags -> 2 x v_wmma per k-step.
// sA: [m][k] row-major (A-frag = two contiguous 16B runs per lane).
// sB: staged TRANSPOSED [n][k]   (B-frag = two contiguous 16B runs per lane).
// ---------------------------------------------------------------------------
#define KTS 32
__launch_bounds__(256)
__global__ void wmma_gemm_kernel(const float* __restrict__ A,
                                 const float* __restrict__ W,
                                 const float* __restrict__ bias,  // may be null
                                 float* __restrict__ C,
                                 int M, int N, int K, int relu) {
  __shared__ alignas(16) bf16_t sA[64 * KTS];   // [m][k]
  __shared__ alignas(16) bf16_t sB[64 * KTS];   // [n][k]  (transposed stage)

  const int tid  = threadIdx.x;
  const int wave = tid >> 5, lane = tid & 31;
  const int wm = wave >> 1, wn = wave & 1;          // 4 x 2 wave grid
  const int mblk = blockIdx.y * 64, nblk = blockIdx.x * 64;
  const int half = lane >> 4;
  const int mrow = lane & 15;                       // A row within tile
  const int ncol = lane & 15;                       // B/C col within tile

  v8f acc0 = {}, acc1 = {};

  for (int kb = 0; kb < K; kb += KTS) {
    // Stage A tile 64x32 as bf16 (float4 global reads, v4bf LDS writes)
#pragma unroll
    for (int it = 0; it < 2; ++it) {
      const int i = tid + it * 256;
      const int r = i >> 3, c4 = (i & 7) * 4;
      const float4 a = *(const float4*)&A[(size_t)(mblk + r) * K + kb + c4];
      v4bf p = { (bf16_t)a.x, (bf16_t)a.y, (bf16_t)a.z, (bf16_t)a.w };
      *(v4bf*)&sA[r * KTS + c4] = p;
      const float4 b = *(const float4*)&W[(size_t)(nblk + r) * K + kb + c4];
      v4bf q = { (bf16_t)b.x, (bf16_t)b.y, (bf16_t)b.z, (bf16_t)b.w };
      *(v4bf*)&sB[r * KTS + c4] = q;
    }
    __syncthreads();

    // Fragments: two ds_load_b128 each (ISA 7.12.2 layouts)
    const bf16_t* ar = &sA[(wm * 16 + mrow) * KTS];
    v16bf af = cat8(*(const v8bf*)&ar[half * 8],
                    *(const v8bf*)&ar[16 + half * 8]);
    const bf16_t* br0 = &sB[(wn * 32 + ncol) * KTS + half * 8];
    v16bf bf0 = cat8(*(const v8bf*)&br0[0], *(const v8bf*)&br0[16]);
    const bf16_t* br1 = &sB[(wn * 32 + 16 + ncol) * KTS + half * 8];
    v16bf bf1 = cat8(*(const v8bf*)&br1[0], *(const v8bf*)&br1[16]);

    acc0 = __builtin_amdgcn_wmma_f32_16x16x32_bf16(false, af, false, bf0,
                                                   (short)0, acc0, false, false);
    acc1 = __builtin_amdgcn_wmma_f32_16x16x32_bf16(false, af, false, bf1,
                                                   (short)0, acc1, false, false);
    __syncthreads();
  }

  // C layout: VGPR d -> row = half*8 + d, col = lane&15
  const int col0 = nblk + wn * 32 + ncol;
  const int col1 = col0 + 16;
  const float bv0 = bias ? bias[col0] : 0.0f;
  const float bv1 = bias ? bias[col1] : 0.0f;
#pragma unroll
  for (int d = 0; d < 8; ++d) {
    const int row = mblk + wm * 16 + half * 8 + d;
    float u0 = acc0[d] + bv0;
    float u1 = acc1[d] + bv1;
    if (relu) {
      u0 = u0 > 0.0f ? u0 : 0.0f;
      u1 = u1 > 0.0f ? u1 : 0.0f;
    }
    C[(size_t)row * N + col0] = u0;
    C[(size_t)row * N + col1] = u1;
  }
}

// ---------------------------------------------------------------------------
// Row LayerNorm in place: X[row, 0..ncols) normalized, then *g + b
// ---------------------------------------------------------------------------
__launch_bounds__(256)
__global__ void rowln_kernel(float* __restrict__ X, const float* __restrict__ g,
                             const float* __restrict__ b, int ncols) {
  __shared__ float red[32];
  float* xr = X + (size_t)blockIdx.x * ncols;
  float s = 0.0f, ss = 0.0f;
  for (int i = threadIdx.x; i < ncols; i += blockDim.x) {
    float v = xr[i];
    s += v;
    ss += v * v;
  }
  s  = block_sum(s, red);
  ss = block_sum(ss, red);
  const float mean = s / (float)ncols;
  const float var  = ss / (float)ncols - mean * mean;
  const float rs   = rsqrtf(var + EPSLN);
  for (int i = threadIdx.x; i < ncols; i += blockDim.x)
    xr[i] = (xr[i] - mean) * rs * g[i] + b[i];
}

// ---------------------------------------------------------------------------
// Grid-wide barrier (persistent kernel; all NWG blocks resident)
// ---------------------------------------------------------------------------
__device__ __forceinline__ void grid_barrier(unsigned int* bar, unsigned int nwg) {
  __syncthreads();
  if (threadIdx.x == 0) {
    __threadfence();
    const unsigned int gen = __atomic_load_n(&bar[1], __ATOMIC_RELAXED);
    const unsigned int t = atomicAdd(&bar[0], 1u);
    if (t == nwg - 1u) {
      __atomic_store_n(&bar[0], 0u, __ATOMIC_RELAXED);
      __atomic_fetch_add(&bar[1], 1u, __ATOMIC_RELEASE);
    } else {
      while (__atomic_load_n(&bar[1], __ATOMIC_ACQUIRE) == gen)
        __builtin_amdgcn_s_sleep(2);
    }
    __threadfence();
  }
  __syncthreads();
}

// ---------------------------------------------------------------------------
// Persistent LayerNorm-LSTM recurrence.
//   NWG = 64 WGs x 1024 threads. Each WG owns 64 of the 4096 gate rows; its
//   w_hh slice (64x1024 f32 = 256KB) lives in LDS for all 8192 steps.
//   phase 1 (all WGs): hg = h @ whh^T, 16 threads/row, float4 LDS reads.
//   phase 2 (all threads of every WG? no -- WG0 only): hidden unit = tid;
//   the thread's four GEMV results ARE its own i/f/g/o gates, so the state
//   update is thread-local apart from the two LN reductions.
// ---------------------------------------------------------------------------
#define NWG 64
#define ROWS_PER_WG 64  // 4096 / NWG
__launch_bounds__(1024, 1)
__global__ void lnlstm_recur_kernel(const float* __restrict__ igates, // [T,4096] (LN'd)
                                    const float* __restrict__ whh,    // [4096,1024]
                                    const float* __restrict__ gh,
                                    const float* __restrict__ bh,
                                    const float* __restrict__ gc,
                                    const float* __restrict__ bc,
                                    float* __restrict__ ys,           // [T,1024]
                                    float* __restrict__ hbuf,         // [1024]
                                    float* __restrict__ hgbuf,        // [4096]
                                    unsigned int* __restrict__ bar,   // [2], zeroed
                                    int T) {
  __shared__ alignas(16) float sW[ROWS_PER_WG * 1024];  // 256 KB resident slice
  __shared__ alignas(16) float sH[1024];
  __shared__ float red[32];

  const int tid = threadIdx.x;
  const int wg  = blockIdx.x;

  // Load this WG's w_hh slice into LDS once (coalesced float4).
  const float4* wsrc = (const float4*)(whh + (size_t)wg * ROWS_PER_WG * 1024);
  float4* wdst = (float4*)sW;
  for (int i = tid; i < ROWS_PER_WG * 256; i += 1024) wdst[i] = wsrc[i];

  float creg = 0.0f;  // cell state for hidden unit `tid` (WG0 only)
  if (wg == 0) hbuf[tid] = 0.0f;

  grid_barrier(bar, NWG);

  const int r  = tid >> 4;   // gate row within slice (0..63)
  const int kc = tid & 15;   // k chunk: float4s kc, kc+16, ..., kc+240

  for (int t = 0; t < T; ++t) {
    // ---- phase 1: hg = h @ w_hh^T (weights from LDS) ----
    sH[tid] = hbuf[tid];
    __syncthreads();

    const float4* w4 = (const float4*)sW + r * 256 + kc;
    const float4* h4 = (const float4*)sH + kc;
    float p = 0.0f;
#pragma unroll
    for (int j = 0; j < 16; ++j) {
      const float4 a = w4[j * 16];
      const float4 h = h4[j * 16];
      p += a.x * h.x + a.y * h.y + a.z * h.z + a.w * h.w;
    }
#pragma unroll
    for (int o = 8; o > 0; o >>= 1) p += __shfl_xor(p, o, 32);
    if (kc == 0) hgbuf[wg * ROWS_PER_WG + r] = p;

    grid_barrier(bar, NWG);

    // ---- phase 2: LN(hg), gates, c/h update (WG0 only) ----
    if (wg == 0) {
      float v[4];
      float s = 0.0f, ssq = 0.0f;
#pragma unroll
      for (int j = 0; j < 4; ++j) {
        v[j] = hgbuf[tid + 1024 * j];
        s += v[j];
        ssq += v[j] * v[j];
      }
      s   = block_sum(s, red);
      ssq = block_sum(ssq, red);
      const float mean = s * (1.0f / 4096.0f);
      const float var  = ssq * (1.0f / 4096.0f) - mean * mean;
      const float rs   = rsqrtf(var + EPSLN);
      const float* igt = igates + (size_t)t * 4096;
      float gate[4];
#pragma unroll
      for (int j = 0; j < 4; ++j) {
        const int idx = tid + 1024 * j;
        gate[j] = (v[j] - mean) * rs * gh[idx] + bh[idx] + igt[idx];
      }
      // gate[0..3] = i, f, g, o for hidden unit `tid`
      const float cn = sigm(gate[1]) * creg + sigm(gate[0]) * tanhf(gate[2]);
      creg = cn;
      float s2  = block_sum(cn, red);
      float ss2 = block_sum(cn * cn, red);
      const float mc  = s2 * (1.0f / 1024.0f);
      const float vc  = ss2 * (1.0f / 1024.0f) - mc * mc;
      const float rsc = rsqrtf(vc + EPSLN);
      const float hn  = sigm(gate[3]) * tanhf((cn - mc) * rsc * gc[tid] + bc[tid]);
      hbuf[tid] = hn;
      ys[(size_t)t * 1024 + tid] = hn;
    }

    grid_barrier(bar, NWG);
  }
}

// ---------------------------------------------------------------------------
// Host launcher
// ---------------------------------------------------------------------------
extern "C" void kernel_launch(void* const* d_in, const int* in_sizes, int n_in,
                              void* d_out, int out_size, void* d_ws, size_t ws_size,
                              hipStream_t stream) {
  (void)in_sizes; (void)n_in; (void)out_size; (void)ws_size;

  constexpr int T = 8192, H = 1024, G4 = 4096, L = 2;

  const float* x     = (const float*)d_in[0];   // [T,1,1024]
  const float* w_ih  = (const float*)d_in[1];   // [L,4096,1024]
  const float* w_hh  = (const float*)d_in[2];   // [L,4096,1024]
  const float* ln_gi = (const float*)d_in[3];   // [L,4096]
  const float* ln_bi = (const float*)d_in[4];
  const float* ln_gh = (const float*)d_in[5];
  const float* ln_bh = (const float*)d_in[6];
  const float* ln_gc = (const float*)d_in[7];   // [L,1024]
  const float* ln_bc = (const float*)d_in[8];
  const float* fc_w  = (const float*)d_in[9];   // [1024,1024]
  const float* fc_b  = (const float*)d_in[10];  // [1024]
  float* out = (float*)d_out;                   // [T,1,1024]

  // workspace layout (f32): igates 128MB | ys 32MB | h 4KB | hg 16KB | bar
  float* igates = (float*)d_ws;
  float* ys     = igates + (size_t)T * G4;
  float* hbuf   = ys + (size_t)T * H;
  float* hgbuf  = hbuf + H;
  unsigned int* bar = (unsigned int*)(hgbuf + G4);

  for (int l = 0; l < L; ++l) {
    const float* seq_in = (l == 0) ? x : ys;
    const float* wih_l  = w_ih + (size_t)l * G4 * H;
    const float* whh_l  = w_hh + (size_t)l * G4 * H;

    // igates = seq_in @ w_ih^T   (M=8192, N=4096, K=1024) via bf16 WMMA
    dim3 g1(G4 / 64, T / 64);
    wmma_gemm_kernel<<<g1, 256, 0, stream>>>(seq_in, wih_l, nullptr, igates,
                                             T, G4, H, 0);
    // LN over each 4096-wide row
    rowln_kernel<<<T, 256, 0, stream>>>(igates, ln_gi + (size_t)l * G4,
                                        ln_bi + (size_t)l * G4, G4);

    // recurrence (persistent, weights in LDS)
    hipMemsetAsync(bar, 0, 2 * sizeof(unsigned int), stream);
    lnlstm_recur_kernel<<<NWG, 1024, 0, stream>>>(
        igates, whh_l, ln_gh + (size_t)l * G4, ln_bh + (size_t)l * G4,
        ln_gc + (size_t)l * H, ln_bc + (size_t)l * H, ys, hbuf, hgbuf, bar, T);
  }

  // FC + ReLU: out = relu(ys @ fc_w^T + fc_b)  (M=8192, N=1024, K=1024)
  dim3 g2(H / 64, T / 64);
  wmma_gemm_kernel<<<g2, 256, 0, stream>>>(ys, fc_w, fc_b, out, T, H, H, 1);
}
